// LongContextAttention_76433237999888
// MI455X (gfx1250) — matrix operand order repaired and checked by reading, not compiled, not run on hardware
//
#include <hip/hip_runtime.h>

// ---- CDNA5 (gfx1250) flash-attention forward, causal, fp32 I/O, f16 WMMA ----
// v3: fix async-builtin pointer types (v4i addrspace(1)/(3) pointers).

typedef __attribute__((ext_vector_type(16))) _Float16 v16h;
typedef __attribute__((ext_vector_type(8)))  _Float16 v8h;
typedef __attribute__((ext_vector_type(8)))  float    v8f;

union V16 { v16h v; struct { v8h lo, hi; } h; };

// async copy operand types: pointer to 16-byte int vector in AS1 / AS3
typedef int v4i __attribute__((vector_size(16)));
typedef __attribute__((address_space(1))) v4i gv4i;   // global
typedef __attribute__((address_space(3))) v4i sv4i;   // LDS

constexpr int BB = 2;     // batch
constexpr int SS = 2048;  // sequence
constexpr int HH = 16;    // heads
constexpr int DD = 128;   // head dim
constexpr int BM = 64;    // query rows per block (4 waves x 16)
constexpr int BC = 32;    // key cols per iteration (== WMMA K for PV)
constexpr int NW = 4;     // waves per block

#if __has_builtin(__builtin_amdgcn_global_load_async_to_lds_b128) && \
    __has_builtin(__builtin_amdgcn_s_wait_asynccnt)
#define HAVE_ASYNC 1
#else
#define HAVE_ASYNC 0
#endif

// lane-XOR exchange: single ds_swizzle (group-of-32, and=0x1F, xor=m) if available
#if __has_builtin(__builtin_amdgcn_ds_swizzle)
#define XRED(x, m) __builtin_bit_cast(float, \
    __builtin_amdgcn_ds_swizzle(__builtin_bit_cast(int, (x)), (0x1f | ((m) << 10))))
#else
#define XRED(x, m) __shfl_xor((x), (m), 32)
#endif

__global__ __launch_bounds__(128, 1)
void fa_fwd_causal(const float* __restrict__ Qg, const float* __restrict__ Kg,
                   const float* __restrict__ Vg, float* __restrict__ Og)
{
    // f16 compute tiles
    __shared__ _Float16 Ksh[BC][DD + 8];     // (key, d), stride 272 B: aligned, conflict-free
    __shared__ _Float16 Vsh[DD][BC + 8];     // transposed (d, key), stride 80 B
    __shared__ _Float16 Psh[NW][16][BC];     // per-wave P relayout staging
#if HAVE_ASYNC
    // raw f32 double buffers fed by the async engine (each thread owns its slice)
    __shared__ float Kraw[2][BC * DD];
    __shared__ float Vraw[2][BC * DD];
#endif

    const int tid  = threadIdx.x;
    const int wave = tid >> 5;
    const int lane = tid & 31;
    const int rlo  = lane & 15;   // column index inside B/C/D fragments
    const int half = lane >> 4;   // which half-wave
    const int rOff = half ? 8 : 0;

    const int bh = blockIdx.y;
    const int b  = bh / HH;
    const int h  = bh % HH;
    const int q0 = blockIdx.x * BM;
    const int qrow = q0 + wave * 16;       // this wave's first query row

    const float scale = 0.08838834764831845f;  // 1/sqrt(128)

    // ---------- Q fragments: A-layout (16x32 f16 per d-chunk), scale folded ----------
    const int klo8 = half ? 8 : 0;
    const float* qp = Qg + (((size_t)b * SS + (qrow + rlo)) * HH + h) * DD;
    v16h aq[4];
#pragma unroll
    for (int c = 0; c < 4; ++c) {
#pragma unroll
        for (int j = 0; j < 8; ++j) {
            aq[c][j]     = (_Float16)(qp[c * 32 + klo8 + j]      * scale);
            aq[c][8 + j] = (_Float16)(qp[c * 32 + klo8 + 16 + j] * scale);
        }
    }

    const v8f vzero = {};
    v8f o[8];
#pragma unroll
    for (int c = 0; c < 8; ++c) o[c] = vzero;
    float mrow[8], lrow[8];
#pragma unroll
    for (int r = 0; r < 8; ++r) { mrow[r] = -1e30f; lrow[r] = 0.0f; }

    // cooperative staging slice: 128 threads cover 32 keys x 128 d
    const int kr = tid >> 2;          // key row 0..31
    const int dc = (tid & 3) * 32;    // d chunk start
    const size_t kvStride = (size_t)HH * DD;            // floats between key rows
    const float* kbase = Kg + (((size_t)b * SS + kr) * HH + h) * DD + dc;
    const float* vbase = Vg + (((size_t)b * SS + kr) * HH + h) * DD + dc;

    const int klo16 = half ? 16 : 0;
    const int nblocks = (q0 + BM - 1) / BC + 1;   // causal bound for whole block

#if HAVE_ASYNC
    // prologue: stream tile 0 into raw buffer 0 (ASYNCcnt-tracked, no VGPR round trip)
    {
        const float* kp = kbase;
        const float* vp = vbase;
        float* kd = &Kraw[0][kr * DD + dc];
        float* vd = &Vraw[0][kr * DD + dc];
#pragma unroll
        for (int t = 0; t < 32; t += 4) {
            __builtin_amdgcn_global_load_async_to_lds_b128(
                (gv4i*)(kp + t), (sv4i*)(kd + t), 0, 0);
            __builtin_amdgcn_global_load_async_to_lds_b128(
                (gv4i*)(vp + t), (sv4i*)(vd + t), 0, 0);
        }
    }
#endif

    for (int j = 0; j < nblocks; ++j) {
        const int k0 = j * BC;

        __syncthreads();   // all waves done reading previous f16 tiles

#if HAVE_ASYNC
        __builtin_amdgcn_s_wait_asynccnt(0);   // my slice of raw tile j has landed
        if (j + 1 < nblocks) {                 // stream tile j+1 while we convert+compute
            const float* kp = kbase + (size_t)(k0 + BC) * kvStride;
            const float* vp = vbase + (size_t)(k0 + BC) * kvStride;
            float* kd = &Kraw[(j + 1) & 1][kr * DD + dc];
            float* vd = &Vraw[(j + 1) & 1][kr * DD + dc];
#pragma unroll
            for (int t = 0; t < 32; t += 4) {
                __builtin_amdgcn_global_load_async_to_lds_b128(
                    (gv4i*)(kp + t), (sv4i*)(kd + t), 0, 0);
                __builtin_amdgcn_global_load_async_to_lds_b128(
                    (gv4i*)(vp + t), (sv4i*)(vd + t), 0, 0);
            }
        }
        const float* ksrc = &Kraw[j & 1][kr * DD + dc];
        const float* vsrc = &Vraw[j & 1][kr * DD + dc];
#else
        const float* ksrc = kbase + (size_t)k0 * kvStride;
        const float* vsrc = vbase + (size_t)k0 * kvStride;
        if (j + 1 < nblocks) {
            __builtin_prefetch(ksrc + BC * kvStride, 0, 1);
            __builtin_prefetch(vsrc + BC * kvStride, 0, 1);
        }
#endif
        // ---------- convert raw f32 -> f16 tiles (K row-major, V transposed) ----------
#pragma unroll
        for (int t = 0; t < 32; t += 4) {
            float4 kf = *(const float4*)(ksrc + t);
            Ksh[kr][dc + t + 0] = (_Float16)kf.x;
            Ksh[kr][dc + t + 1] = (_Float16)kf.y;
            Ksh[kr][dc + t + 2] = (_Float16)kf.z;
            Ksh[kr][dc + t + 3] = (_Float16)kf.w;
            float4 vf = *(const float4*)(vsrc + t);
            Vsh[dc + t + 0][kr] = (_Float16)vf.x;
            Vsh[dc + t + 1][kr] = (_Float16)vf.y;
            Vsh[dc + t + 2][kr] = (_Float16)vf.z;
            Vsh[dc + t + 3][kr] = (_Float16)vf.w;
        }
        __syncthreads();   // f16 tiles visible to all waves

        // causal early-out for this wave (barriers above keep counts uniform)
        if (k0 > qrow + 15) continue;

        // ---------- scores: S = (Q*scale) . K^T  (16 x 32, two 16x16 C-frags) ----------
        v8f s0 = vzero, s1 = vzero;
#pragma unroll
        for (int c = 0; c < 4; ++c) {
            V16 kb0, kb1;
            kb0.h.lo = *(const v8h*)&Ksh[rlo][c * 32 + klo8];
            kb0.h.hi = *(const v8h*)&Ksh[rlo][c * 32 + klo8 + 16];
            s0 = __builtin_amdgcn_wmma_f32_16x16x32_f16(false, aq[c], false, kb0.v,
                                                        (short)0, s0, false, false);
            kb1.h.lo = *(const v8h*)&Ksh[16 + rlo][c * 32 + klo8];
            kb1.h.hi = *(const v8h*)&Ksh[16 + rlo][c * 32 + klo8 + 16];
            s1 = __builtin_amdgcn_wmma_f32_16x16x32_f16(false, aq[c], false, kb1.v,
                                                        (short)0, s1, false, false);
        }

        // ---------- online softmax; mask only on diagonal blocks ----------
        const bool diag = (k0 + BC - 1) > qrow;   // wave-uniform
        if (diag) {
#pragma unroll
            for (int r = 0; r < 8; ++r) {
                const int grow = qrow + r + rOff;
                float e0 = (k0 + rlo      <= grow) ? s0[r] : -1e30f;
                float e1 = (k0 + 16 + rlo <= grow) ? s1[r] : -1e30f;
                float t = fmaxf(e0, e1);
                t = fmaxf(t, XRED(t, 1));
                t = fmaxf(t, XRED(t, 2));
                t = fmaxf(t, XRED(t, 4));
                t = fmaxf(t, XRED(t, 8));
                const float mnew  = fmaxf(mrow[r], t);
                const float alpha = __expf(mrow[r] - mnew);
                const float p0 = __expf(e0 - mnew);
                const float p1 = __expf(e1 - mnew);
                float rs = p0 + p1;
                rs += XRED(rs, 1);
                rs += XRED(rs, 2);
                rs += XRED(rs, 4);
                rs += XRED(rs, 8);
                lrow[r] = lrow[r] * alpha + rs;
                mrow[r] = mnew;
#pragma unroll
                for (int c = 0; c < 8; ++c) o[c][r] *= alpha;
                Psh[wave][r + rOff][rlo]      = (_Float16)p0;
                Psh[wave][r + rOff][16 + rlo] = (_Float16)p1;
            }
        } else {
#pragma unroll
            for (int r = 0; r < 8; ++r) {
                const float e0 = s0[r];
                const float e1 = s1[r];
                float t = fmaxf(e0, e1);
                t = fmaxf(t, XRED(t, 1));
                t = fmaxf(t, XRED(t, 2));
                t = fmaxf(t, XRED(t, 4));
                t = fmaxf(t, XRED(t, 8));
                const float mnew  = fmaxf(mrow[r], t);
                const float alpha = __expf(mrow[r] - mnew);
                const float p0 = __expf(e0 - mnew);
                const float p1 = __expf(e1 - mnew);
                float rs = p0 + p1;
                rs += XRED(rs, 1);
                rs += XRED(rs, 2);
                rs += XRED(rs, 4);
                rs += XRED(rs, 8);
                lrow[r] = lrow[r] * alpha + rs;
                mrow[r] = mnew;
#pragma unroll
                for (int c = 0; c < 8; ++c) o[c][r] *= alpha;
                Psh[wave][r + rOff][rlo]      = (_Float16)p0;
                Psh[wave][r + rOff][16 + rlo] = (_Float16)p1;
            }
        }

        // ---------- PV: O += P . V   (A = P via per-wave LDS relayout) ----------
        V16 pa;
        pa.h.lo = *(const v8h*)&Psh[wave][rlo][klo8];
        pa.h.hi = *(const v8h*)&Psh[wave][rlo][klo8 + 16];
#pragma unroll
        for (int c = 0; c < 8; ++c) {
            V16 vb;
            vb.h.lo = *(const v8h*)&Vsh[c * 16 + rlo][klo8];
            vb.h.hi = *(const v8h*)&Vsh[c * 16 + rlo][klo8 + 16];
            o[c] = __builtin_amdgcn_wmma_f32_16x16x32_f16(false, pa.v, false, vb.v,
                                                          (short)0, o[c], false, false);
        }
    }

    // ---------- epilogue: normalize and store (C-layout -> [b,q,h,d]) ----------
#pragma unroll
    for (int r = 0; r < 8; ++r) {
        const float inv = 1.0f / lrow[r];
        float* orow = Og + (((size_t)b * SS + (qrow + r + rOff)) * HH + h) * DD;
#pragma unroll
        for (int c = 0; c < 8; ++c)
            orow[c * 16 + rlo] = o[c][r] * inv;
    }
}

extern "C" void kernel_launch(void* const* d_in, const int* in_sizes, int n_in,
                              void* d_out, int out_size, void* d_ws, size_t ws_size,
                              hipStream_t stream) {
    const float* Q = (const float*)d_in[0];
    const float* K = (const float*)d_in[1];
    const float* V = (const float*)d_in[2];
    float* O = (float*)d_out;

    dim3 grid(SS / BM, BB * HH);   // 32 x 32 blocks
    dim3 block(128);               // 4 waves (wave32)
    fa_fwd_causal<<<grid, block, 0, stream>>>(Q, K, V, O);
}
